// MoMuGNN_29755533427121
// MI455X (gfx1250) — compile-verified, matrix operand order checked by reading, not run on hardware
//
#include <hip/hip_runtime.h>
#include <stdint.h>

#define N_NODES 50000
#define N_EDGES 400000
#define DIM 256
#define HID 512
#define N_LAYERS 5
#define BN_EPS 1e-5f
#define MTILES 3125          // N_NODES / 16 (exact)
#define KT1 8                // DIM / 32
#define KT2 16               // HID / 32

typedef __attribute__((ext_vector_type(16))) __bf16 v16bf;
typedef __attribute__((ext_vector_type(8)))  float  v8f;

union Frag { v16bf v; uint4 q[2]; };

__device__ __forceinline__ unsigned short f2bf(float x) {
    union { __bf16 b; unsigned short s; } c;
    c.b = (__bf16)x;                       // hardware RNE f32->bf16
    return c.s;
}

// ushort offset of element (m, kin) inside a 16x32 bf16 fragment tile (512 elems, lane-major)
__device__ __forceinline__ int swz_off(int m, int kin) {
    int hi  = kin >> 4;
    int l16 = (kin >> 3) & 1;
    int p   = kin & 7;
    int r   = hi * 4 + (p >> 1);
    int j   = r * 2 + (p & 1);
    return (l16 * 16 + m) * 16 + j;
}

// ---- one-time: W[l][k][n] (f32) -> fragment-swizzled bf16 tiles [l][n/16][k/32] ----
__global__ void k_swz_w(const float* __restrict__ in, unsigned short* __restrict__ out,
                        int K, int Nc, int total) {
    int idx = blockIdx.x * 256 + threadIdx.x;
    if (idx >= total) return;
    int per = K * Nc;
    int l = idx / per;
    int rem = idx - l * per;
    int k = rem / Nc;
    int n = rem - k * Nc;
    int tile = (n >> 4) * (K >> 5) + (k >> 5);
    out[(size_t)l * per + (size_t)tile * 512 + swz_off(n & 15, k & 31)] = f2bf(in[idx]);
}

// ---- per-layer: agg[N,256] f32 -> fragment-swizzled bf16 (A-operand for GEMM1) ----
__global__ void k_swz_agg(const float* __restrict__ A, unsigned short* __restrict__ out) {
    int idx = blockIdx.x * 256 + threadIdx.x;      // covers N_NODES*DIM exactly
    int row = idx >> 8;
    int k = idx & 255;
    int tile = (row >> 4) * KT1 + (k >> 5);
    out[(size_t)tile * 512 + swz_off(row & 15, k & 31)] = f2bf(A[idx]);
}

// ---- agg init with self-loop: agg[i,c] = h[i,c] + E1[l][0][c] + E2[l][0][c] ----
__global__ void k_init_agg(const float* __restrict__ h, const float* __restrict__ e1,
                           const float* __restrict__ e2, float* __restrict__ agg) {
    int idx = blockIdx.x * 256 + threadIdx.x;
    int ch = idx & (DIM - 1);
    agg[idx] = h[idx] + e1[ch] + e2[ch];
}

// ---- edge scatter: one wave per edge, lane owns 8 channels ----
__global__ void k_edge_scatter(const float* __restrict__ h, const int* __restrict__ ei,
                               const int* __restrict__ ea, const float* __restrict__ e1,
                               const float* __restrict__ e2, float* __restrict__ agg) {
    int lane = threadIdx.x & 31;
    int wave = blockIdx.x * (blockDim.x >> 5) + (threadIdx.x >> 5);
    int nwaves = gridDim.x * (blockDim.x >> 5);
    int c0 = lane * 8;
    for (int e = wave; e < N_EDGES; e += nwaves) {
        int s  = ei[e];
        int d  = ei[N_EDGES + e];
        int a0 = ea[2 * e];
        int a1 = ea[2 * e + 1];
        int en = e + nwaves;
        if (en < N_EDGES) {                       // hide latency of next random gather
            int sn = ei[en];
            __builtin_prefetch(h + (size_t)sn * DIM + c0, 0, 3);
        }
        const float4* hp = (const float4*)(h  + (size_t)s  * DIM + c0);
        const float4* p1 = (const float4*)(e1 + (size_t)a0 * DIM + c0);
        const float4* p2 = (const float4*)(e2 + (size_t)a1 * DIM + c0);
        float4 h0 = hp[0], h1 = hp[1];
        float4 x0 = p1[0], x1 = p1[1];
        float4 y0 = p2[0], y1 = p2[1];
        float* ap = agg + (size_t)d * DIM + c0;
        atomicAdd(ap + 0, h0.x + x0.x + y0.x);
        atomicAdd(ap + 1, h0.y + x0.y + y0.y);
        atomicAdd(ap + 2, h0.z + x0.z + y0.z);
        atomicAdd(ap + 3, h0.w + x0.w + y0.w);
        atomicAdd(ap + 4, h1.x + x1.x + y1.x);
        atomicAdd(ap + 5, h1.y + x1.y + y1.y);
        atomicAdd(ap + 6, h1.z + x1.z + y1.z);
        atomicAdd(ap + 7, h1.w + x1.w + y1.w);
    }
}

// ---- GEMM1: tmpB(swz)[N,512] = bf16(relu(aggB(swz)[N,256] @ W1 + b1)) ----
// block = 64 rows x 128 cols, 8 waves, each wave a 32x32 output tile (2x2 WMMA tiles)
__global__ void __launch_bounds__(256)
k_gemm1(const unsigned short* __restrict__ Aw, const unsigned short* __restrict__ Bw,
        const float* __restrict__ bias, unsigned short* __restrict__ C) {
    int lane = threadIdx.x & 31;
    int w    = threadIdx.x >> 5;
    int l16 = lane >> 4;
    int lm  = lane & 15;
    int wm = w & 1, wn = w >> 1;
    int rowBase = blockIdx.x * 64 + wm * 32;
    int colBase = blockIdx.y * 128 + wn * 32;
    int ti0 = rowBase >> 4, ti1 = ti0 + 1;
    if (ti0 >= MTILES) ti0 = 0;                  // D rows independent in M; stores guarded
    if (ti1 >= MTILES) ti1 = 0;
    int nj0 = colBase >> 4, nj1 = nj0 + 1;
    v8f acc00 = {0.f,0.f,0.f,0.f,0.f,0.f,0.f,0.f};
    v8f acc01 = {0.f,0.f,0.f,0.f,0.f,0.f,0.f,0.f};
    v8f acc10 = {0.f,0.f,0.f,0.f,0.f,0.f,0.f,0.f};
    v8f acc11 = {0.f,0.f,0.f,0.f,0.f,0.f,0.f,0.f};
#pragma unroll
    for (int tk = 0; tk < KT1; ++tk) {
        Frag a0, a1, b0, b1;
        const uint4* ap0 = (const uint4*)(Aw + (size_t)(ti0 * KT1 + tk) * 512);
        const uint4* ap1 = (const uint4*)(Aw + (size_t)(ti1 * KT1 + tk) * 512);
        const uint4* bp0 = (const uint4*)(Bw + (size_t)(nj0 * KT1 + tk) * 512);
        const uint4* bp1 = (const uint4*)(Bw + (size_t)(nj1 * KT1 + tk) * 512);
        a0.q[0] = ap0[lane * 2]; a0.q[1] = ap0[lane * 2 + 1];
        a1.q[0] = ap1[lane * 2]; a1.q[1] = ap1[lane * 2 + 1];
        b0.q[0] = bp0[lane * 2]; b0.q[1] = bp0[lane * 2 + 1];
        b1.q[0] = bp1[lane * 2]; b1.q[1] = bp1[lane * 2 + 1];
        acc00 = __builtin_amdgcn_wmma_f32_16x16x32_bf16(false, a0.v, false, b0.v, (short)0, acc00, false, false);
        acc01 = __builtin_amdgcn_wmma_f32_16x16x32_bf16(false, a0.v, false, b1.v, (short)0, acc01, false, false);
        acc10 = __builtin_amdgcn_wmma_f32_16x16x32_bf16(false, a1.v, false, b0.v, (short)0, acc10, false, false);
        acc11 = __builtin_amdgcn_wmma_f32_16x16x32_bf16(false, a1.v, false, b1.v, (short)0, acc11, false, false);
    }
    float bb0 = bias[colBase + lm];
    float bb1 = bias[colBase + 16 + lm];
    int kc0 = colBase + lm, kc1 = colBase + 16 + lm;
#pragma unroll
    for (int r = 0; r < 8; ++r) {
        int m0 = rowBase + l16 * 8 + r;
        int m1 = m0 + 16;
        if (m0 < N_NODES) {
            float v0 = acc00[r] + bb0; v0 = v0 > 0.f ? v0 : 0.f;
            float v1 = acc01[r] + bb1; v1 = v1 > 0.f ? v1 : 0.f;
            int mm = m0 & 15, mt = m0 >> 4;
            C[(size_t)(mt * KT2 + (kc0 >> 5)) * 512 + swz_off(mm, kc0 & 31)] = f2bf(v0);
            C[(size_t)(mt * KT2 + (kc1 >> 5)) * 512 + swz_off(mm, kc1 & 31)] = f2bf(v1);
        }
        if (m1 < N_NODES) {
            float v0 = acc10[r] + bb0; v0 = v0 > 0.f ? v0 : 0.f;
            float v1 = acc11[r] + bb1; v1 = v1 > 0.f ? v1 : 0.f;
            int mm = m1 & 15, mt = m1 >> 4;
            C[(size_t)(mt * KT2 + (kc0 >> 5)) * 512 + swz_off(mm, kc0 & 31)] = f2bf(v0);
            C[(size_t)(mt * KT2 + (kc1 >> 5)) * 512 + swz_off(mm, kc1 & 31)] = f2bf(v1);
        }
    }
}

// ---- GEMM2: h[N,256] f32 = tmpB(swz)[N,512] @ W2 + b2 ----
__global__ void __launch_bounds__(256)
k_gemm2(const unsigned short* __restrict__ Aw, const unsigned short* __restrict__ Bw,
        const float* __restrict__ bias, float* __restrict__ C) {
    int lane = threadIdx.x & 31;
    int w    = threadIdx.x >> 5;
    int l16 = lane >> 4;
    int lm  = lane & 15;
    int wm = w & 1, wn = w >> 1;
    int rowBase = blockIdx.x * 64 + wm * 32;
    int colBase = blockIdx.y * 128 + wn * 32;
    int ti0 = rowBase >> 4, ti1 = ti0 + 1;
    if (ti0 >= MTILES) ti0 = 0;
    if (ti1 >= MTILES) ti1 = 0;
    int nj0 = colBase >> 4, nj1 = nj0 + 1;
    v8f acc00 = {0.f,0.f,0.f,0.f,0.f,0.f,0.f,0.f};
    v8f acc01 = {0.f,0.f,0.f,0.f,0.f,0.f,0.f,0.f};
    v8f acc10 = {0.f,0.f,0.f,0.f,0.f,0.f,0.f,0.f};
    v8f acc11 = {0.f,0.f,0.f,0.f,0.f,0.f,0.f,0.f};
#pragma unroll
    for (int tk = 0; tk < KT2; ++tk) {
        Frag a0, a1, b0, b1;
        const uint4* ap0 = (const uint4*)(Aw + (size_t)(ti0 * KT2 + tk) * 512);
        const uint4* ap1 = (const uint4*)(Aw + (size_t)(ti1 * KT2 + tk) * 512);
        const uint4* bp0 = (const uint4*)(Bw + (size_t)(nj0 * KT2 + tk) * 512);
        const uint4* bp1 = (const uint4*)(Bw + (size_t)(nj1 * KT2 + tk) * 512);
        a0.q[0] = ap0[lane * 2]; a0.q[1] = ap0[lane * 2 + 1];
        a1.q[0] = ap1[lane * 2]; a1.q[1] = ap1[lane * 2 + 1];
        b0.q[0] = bp0[lane * 2]; b0.q[1] = bp0[lane * 2 + 1];
        b1.q[0] = bp1[lane * 2]; b1.q[1] = bp1[lane * 2 + 1];
        acc00 = __builtin_amdgcn_wmma_f32_16x16x32_bf16(false, a0.v, false, b0.v, (short)0, acc00, false, false);
        acc01 = __builtin_amdgcn_wmma_f32_16x16x32_bf16(false, a0.v, false, b1.v, (short)0, acc01, false, false);
        acc10 = __builtin_amdgcn_wmma_f32_16x16x32_bf16(false, a1.v, false, b0.v, (short)0, acc10, false, false);
        acc11 = __builtin_amdgcn_wmma_f32_16x16x32_bf16(false, a1.v, false, b1.v, (short)0, acc11, false, false);
    }
    float bb0 = bias[colBase + lm];
    float bb1 = bias[colBase + 16 + lm];
#pragma unroll
    for (int r = 0; r < 8; ++r) {
        int m0 = rowBase + l16 * 8 + r;
        int m1 = m0 + 16;
        if (m0 < N_NODES) {
            C[(size_t)m0 * DIM + colBase + lm]      = acc00[r] + bb0;
            C[(size_t)m0 * DIM + colBase + 16 + lm] = acc01[r] + bb1;
        }
        if (m1 < N_NODES) {
            C[(size_t)m1 * DIM + colBase + lm]      = acc10[r] + bb0;
            C[(size_t)m1 * DIM + colBase + 16 + lm] = acc11[r] + bb1;
        }
    }
}

// ---- BatchNorm ----
__global__ void k_zero_stats(float* stats) { stats[threadIdx.x] = 0.f; }

__global__ void k_bn_stats(const float* __restrict__ h, float* __restrict__ stats) {
    int ch = threadIdx.x;
    float s = 0.f, s2 = 0.f;
    for (int r = blockIdx.x; r < N_NODES; r += gridDim.x) {
        float v = h[(size_t)r * DIM + ch];
        s += v; s2 += v * v;
    }
    atomicAdd(&stats[ch], s);
    atomicAdd(&stats[DIM + ch], s2);
}

__global__ void k_bn_apply(const float* __restrict__ hin, float* __restrict__ hout,
                           const float* __restrict__ stats, const float* __restrict__ gamma,
                           const float* __restrict__ beta, int doRelu) {
    int idx = blockIdx.x * 256 + threadIdx.x;
    int ch = idx & (DIM - 1);
    const float inv_n = 1.0f / (float)N_NODES;
    float mu  = stats[ch] * inv_n;
    float var = stats[DIM + ch] * inv_n - mu * mu;
    float y = (hin[idx] - mu) * rsqrtf(var + BN_EPS) * gamma[ch] + beta[ch];
    if (doRelu) y = fmaxf(y, 0.f);
    hout[idx] = y;
}

extern "C" void kernel_launch(void* const* d_in, const int* in_sizes, int n_in,
                              void* d_out, int out_size, void* d_ws, size_t ws_size,
                              hipStream_t stream) {
    (void)in_sizes; (void)n_in; (void)out_size; (void)ws_size;
    const float* x     = (const float*)d_in[0];
    const int*   ei    = (const int*)d_in[1];
    const int*   ea    = (const int*)d_in[2];
    const float* W1    = (const float*)d_in[4];
    const float* b1    = (const float*)d_in[5];
    const float* W2    = (const float*)d_in[6];
    const float* b2    = (const float*)d_in[7];
    const float* E1    = (const float*)d_in[8];
    const float* E2    = (const float*)d_in[9];
    const float* gamma = (const float*)d_in[10];
    const float* beta  = (const float*)d_in[11];
    float* out = (float*)d_out;

    char* ws = (char*)d_ws;
    float* agg           = (float*)ws;          ws += (size_t)N_NODES * DIM * 4;
    float* h             = (float*)ws;          ws += (size_t)N_NODES * DIM * 4;
    unsigned short* aggB = (unsigned short*)ws; ws += (size_t)N_NODES * DIM * 2;
    unsigned short* tmpB = (unsigned short*)ws; ws += (size_t)N_NODES * HID * 2;
    unsigned short* W1s  = (unsigned short*)ws; ws += (size_t)N_LAYERS * DIM * HID * 2;
    unsigned short* W2s  = (unsigned short*)ws; ws += (size_t)N_LAYERS * DIM * HID * 2;
    float* stats         = (float*)ws;          ws += 2 * DIM * 4;

    const int wTotal = N_LAYERS * DIM * HID;
    k_swz_w<<<(wTotal + 255) / 256, 256, 0, stream>>>(W1, W1s, DIM, HID, wTotal);
    k_swz_w<<<(wTotal + 255) / 256, 256, 0, stream>>>(W2, W2s, HID, DIM, wTotal);

    const int mBlocks = (N_NODES + 63) / 64;   // 782
    for (int l = 0; l < N_LAYERS; ++l) {
        const float* hin = (l == 0) ? x : h;
        const float* e1 = E1 + (size_t)l * 6 * DIM;
        const float* e2 = E2 + (size_t)l * 3 * DIM;
        k_init_agg<<<N_NODES, 256, 0, stream>>>(hin, e1, e2, agg);
        k_edge_scatter<<<2048, 256, 0, stream>>>(hin, ei, ea, e1, e2, agg);
        k_swz_agg<<<N_NODES, 256, 0, stream>>>(agg, aggB);
        k_gemm1<<<dim3(mBlocks, HID / 128), 256, 0, stream>>>(aggB, W1s + (size_t)l * DIM * HID,
                                                              b1 + (size_t)l * HID, tmpB);
        k_gemm2<<<dim3(mBlocks, DIM / 128), 256, 0, stream>>>(tmpB, W2s + (size_t)l * DIM * HID,
                                                              b2 + (size_t)l * DIM, h);
        k_zero_stats<<<1, 2 * DIM, 0, stream>>>(stats);
        k_bn_stats<<<250, DIM, 0, stream>>>(h, stats);
        float* hout = (l == N_LAYERS - 1) ? out : h;
        k_bn_apply<<<N_NODES, 256, 0, stream>>>(h, hout, stats,
                                                gamma + (size_t)l * DIM, beta + (size_t)l * DIM,
                                                (l < N_LAYERS - 1) ? 1 : 0);
    }
}
